// LQFormer_9792525435202
// MI455X (gfx1250) — matrix operand order repaired and checked by reading, not compiled, not run on hardware
//
#include <hip/hip_runtime.h>
#include <math.h>
#include <stdint.h>

typedef __attribute__((ext_vector_type(16))) __bf16 v16bf;
typedef __attribute__((ext_vector_type(8)))  __bf16 v8bf;
typedef __attribute__((ext_vector_type(4)))  __bf16 v4bf;
typedef __attribute__((ext_vector_type(2)))  __bf16 v2bf;
typedef __attribute__((ext_vector_type(8)))  float  v8f;
typedef __attribute__((ext_vector_type(2)))  float  v2f;

#define WMMA_BF16(a, b, c) \
  __builtin_amdgcn_wmma_f32_16x16x32_bf16(false, (a), false, (b), (short)0, (c), false, false)

// ---------------- WMMA fragment loaders (wave32, contiguous ds_load_b128 paths) ----------------

// A operand 16x32 bf16 from row-major [row][k] LDS tile (stride: multiple of 8 halves).
__device__ __forceinline__ v16bf frag_a16(const __bf16* base, int stride) {
  const int lane = threadIdx.x & 31;
  const __bf16* p = base + (lane & 15) * stride + ((lane & 16) >> 1);
  const v8bf lo = *(const v8bf*)(p);
  const v8bf hi = *(const v8bf*)(p + 16);
  return __builtin_shufflevector(lo, hi, 0, 1, 2, 3, 4, 5, 6, 7, 8, 9, 10, 11, 12, 13, 14, 15);
}

// B operand 32x16 bf16 from [n][k] row-major LDS tile (B^T storage; stride mult of 8).
__device__ __forceinline__ v16bf frag_b16(const __bf16* base, int stride) {
  const int lane = threadIdx.x & 31;
  const __bf16* p = base + (lane & 15) * stride + ((lane & 16) >> 1);
  const v8bf lo = *(const v8bf*)(p);
  const v8bf hi = *(const v8bf*)(p + 16);
  return __builtin_shufflevector(lo, hi, 0, 1, 2, 3, 4, 5, 6, 7, 8, 9, 10, 11, 12, 13, 14, 15);
}

// gfx1250 async global->LDS copy, 16 bytes per lane (ASYNCcnt-tracked)
__device__ __forceinline__ void async_ld_b128(const __bf16* gaddr, const __bf16* lds_ptr) {
  const unsigned lds = (unsigned)(uintptr_t)lds_ptr;
  asm volatile("global_load_async_to_lds_b128 %0, %1, off" :: "v"(lds), "v"(gaddr) : "memory");
}
__device__ __forceinline__ void wait_async0() {
  asm volatile("s_wait_asynccnt 0" ::: "memory");
}

// ---------------- Generic GEMM: C[M,N] = A[M,K] @ W[K,N] + bias (+resid) ----------------
// Block tile 128x64x32, 8 waves x (2x2) 16x16 tiles, double-buffered LDS + register staging.

template <int OUT_BF16, int RESID>
__global__ __launch_bounds__(256) void gemm_kernel(
    const float* __restrict__ A, const float* __restrict__ W,
    const float* __restrict__ bias, const float* __restrict__ resid,
    void* __restrict__ Cout, int M, int N, int K)
{
  constexpr int BM = 128, BN = 64, BK = 32;
  constexpr int AS = 40, BTS = 40;   // halves; mult of 8 -> 16B aligned fragment rows
  __shared__ __attribute__((aligned(16))) __bf16 Asm[2][BM * AS];    // [row][k]
  __shared__ __attribute__((aligned(16))) __bf16 BsmT[2][BN * BTS];  // [n][k] (transposed)

  const int bm   = blockIdx.y * BM;
  const int bn   = blockIdx.x * BN;
  const int tid  = threadIdx.x;
  const int lane = tid & 31;
  const int wave = tid >> 5;
  const int mt   = (wave & 3) << 1;   // m-tiles mt, mt+1 (of 8)
  const int nt   = (wave >> 2) << 1;  // n-tiles nt, nt+1 (of 4)

  v2f   aReg[8];
  float w0Reg[4], w1Reg[4];

  // Issue next tile's global loads into registers (no LDS dependency -> overlaps with WMMA)
  auto stage_issue = [&](int k0) {
#pragma unroll
    for (int i = 0; i < 8; ++i) {
      const int idx = tid + i * 256;          // 0..2047
      const int row = idx >> 4;               // 0..127
      const int kp  = idx & 15;               // k pair
      const int gr  = min(bm + row, M - 1);   // clamp: garbage rows never stored
      aReg[i] = *(const v2f*)(A + (size_t)gr * K + (k0 + 2 * kp));
    }
#pragma unroll
    for (int i = 0; i < 4; ++i) {
      const int idx = tid + i * 256;          // 0..1023
      const int c   = idx & 63;
      const int rp  = idx >> 6;
      w0Reg[i] = W[(size_t)(k0 + 2 * rp) * N + (bn + c)];
      w1Reg[i] = W[(size_t)(k0 + 2 * rp + 1) * N + (bn + c)];
    }
  };
  // Convert + commit staged registers into LDS buffer
  auto stage_commit = [&](int buf) {
#pragma unroll
    for (int i = 0; i < 8; ++i) {
      const int idx = tid + i * 256;
      const int row = idx >> 4;
      const int kp  = idx & 15;
      v2bf pk; pk[0] = (__bf16)aReg[i][0]; pk[1] = (__bf16)aReg[i][1];
      *(v2bf*)&Asm[buf][row * AS + 2 * kp] = pk;
    }
#pragma unroll
    for (int i = 0; i < 4; ++i) {
      const int idx = tid + i * 256;
      const int c   = idx & 63;
      const int rp  = idx >> 6;
      v2bf pk; pk[0] = (__bf16)w0Reg[i]; pk[1] = (__bf16)w1Reg[i];
      *(v2bf*)&BsmT[buf][c * BTS + 2 * rp] = pk;
    }
  };

  v8f acc00 = {}; v8f acc01 = {}; v8f acc10 = {}; v8f acc11 = {};

  stage_issue(0);
  stage_commit(0);
  __syncthreads();

  const int nk = K / BK;
  for (int kt = 0; kt < nk; ++kt) {
    const int  cur  = kt & 1;
    const bool more = (kt + 1) < nk;
    if (more) stage_issue((kt + 1) * BK);    // global loads in flight during WMMA
    const v16bf a0 = frag_a16(&Asm[cur][(mt * 16) * AS], AS);
    const v16bf a1 = frag_a16(&Asm[cur][((mt + 1) * 16) * AS], AS);
    const v16bf b0 = frag_b16(&BsmT[cur][(nt * 16) * BTS], BTS);
    const v16bf b1 = frag_b16(&BsmT[cur][((nt + 1) * 16) * BTS], BTS);
    acc00 = WMMA_BF16(a0, b0, acc00);
    acc01 = WMMA_BF16(a0, b1, acc01);
    acc10 = WMMA_BF16(a1, b0, acc10);
    acc11 = WMMA_BF16(a1, b1, acc11);
    if (more) stage_commit(cur ^ 1);
    __syncthreads();
  }

  const int col0 = bn + nt * 16 + (lane & 15);
  const int col1 = col0 + 16;
  const float bi0 = bias[col0];
  const float bi1 = bias[col1];
  const int rb0 = bm + mt * 16 + ((lane & 16) >> 1);
  const int rb1 = rb0 + 16;

  auto store_row = [&](int row, float v0, float v1) {
    if constexpr (OUT_BF16) {
      ((__bf16*)Cout)[(size_t)row * N + col0] = (__bf16)v0;
      ((__bf16*)Cout)[(size_t)row * N + col1] = (__bf16)v1;
    } else {
      if constexpr (RESID) {
        v0 += resid[(size_t)row * N + col0];
        v1 += resid[(size_t)row * N + col1];
      }
      ((float*)Cout)[(size_t)row * N + col0] = v0;
      ((float*)Cout)[(size_t)row * N + col1] = v1;
    }
  };

  if (bm + BM <= M) {  // uniform fast path, no guards
#pragma unroll
    for (int r = 0; r < 8; ++r) {
      store_row(rb0 + r, acc00[r] + bi0, acc01[r] + bi1);
      store_row(rb1 + r, acc10[r] + bi0, acc11[r] + bi1);
    }
  } else {
#pragma unroll
    for (int r = 0; r < 8; ++r) {
      if (rb0 + r < M) store_row(rb0 + r, acc00[r] + bi0, acc01[r] + bi1);
      if (rb1 + r < M) store_row(rb1 + r, acc10[r] + bi0, acc11[r] + bi1);
    }
  }
}

// ---------------- In-place LayerNorm over last dim ----------------

__global__ __launch_bounds__(256) void ln_kernel(
    float* __restrict__ x, const float* __restrict__ g, const float* __restrict__ bta, int D)
{
  __shared__ float red[16];
  const size_t base = (size_t)blockIdx.x * D;
  float s = 0.0f, s2 = 0.0f;
  for (int i = threadIdx.x; i < D; i += 256) {
    const float v = x[base + i];
    s += v; s2 += v * v;
  }
#pragma unroll
  for (int off = 16; off >= 1; off >>= 1) {
    s  += __shfl_xor(s,  off, 32);
    s2 += __shfl_xor(s2, off, 32);
  }
  const int wave = threadIdx.x >> 5, lane = threadIdx.x & 31;
  if (lane == 0) { red[wave] = s; red[8 + wave] = s2; }
  __syncthreads();
  if (threadIdx.x == 0) {
    float ts = 0.0f, ts2 = 0.0f;
#pragma unroll
    for (int i = 0; i < 8; ++i) { ts += red[i]; ts2 += red[8 + i]; }
    const float mean = ts / (float)D;
    const float var  = ts2 / (float)D - mean * mean;
    red[0] = mean;
    red[1] = rsqrtf(var + 1e-5f);
  }
  __syncthreads();
  const float mean = red[0], rstd = red[1];
  for (int i = threadIdx.x; i < D; i += 256)
    x[base + i] = (x[base + i] - mean) * rstd * g[i] + bta[i];
}

// ---------------- Flash-style cross-attention (Q=64, NH=8, dqk=72 padded to 96) ----------------

template <int DV, bool MASKED>
__global__ __launch_bounds__(128) void attn_kernel(
    const __bf16* __restrict__ Qp,   // [B, 64, 8*72]
    const __bf16* __restrict__ Kp,   // [B, T, 8*72]
    const __bf16* __restrict__ Vp,   // [B, T, 8*DV]
    const int*    __restrict__ mask, // [B, T] or null
    float*        __restrict__ Out,  // [B, 64, 8*DV]
    int T, float scale)
{
  constexpr int NH = 8, DK = 72, DKP = 96;
  constexpr int NVT = (DV + 15) / 16;
  constexpr int DVP = NVT * 16;
  constexpr int QS = 104, KS = 104;   // halves, mult of 8
  constexpr int NCHV = DVP / 4;
  __shared__ __attribute__((aligned(16))) __bf16 Qs[64 * QS];     // [q][k]
  __shared__ __attribute__((aligned(16))) __bf16 Ks[32 * KS];     // [token][k]
  __shared__ __attribute__((aligned(16))) __bf16 VsT[DVP * 32];   // [feat][token] (transposed)
  __shared__ __attribute__((aligned(16))) __bf16 Ps[4][16 * 40];  // per-wave P tile [row][k]
  __shared__ float Ma[32];

  const int b    = blockIdx.x >> 3;
  const int h    = blockIdx.x & 7;
  const int tid  = threadIdx.x;
  const int lane = tid & 31;
  const int wave = tid >> 5;           // wave owns query rows 16*wave..16*wave+15
  const int hi   = (lane & 16) >> 1;   // row offset 0 or 8 within D-fragment
  const v4bf vz = {};

  const __bf16* qbase = Qp + (size_t)b * 64 * (NH * DK) + h * DK;
  const __bf16* kbase = Kp + (size_t)b * T * (NH * DK) + h * DK;
  const __bf16* vbase = Vp + (size_t)b * T * (NH * DV) + h * DV;

  // Pre-zero the k-padding (cols 72..95) of Qs and Ks once; async copies never touch it.
  for (int i = tid; i < 64 * 6; i += 128) {
    const int r = i / 6, ch = 18 + (i % 6);
    *(v4bf*)&Qs[r * QS + ch * 4] = vz;
  }
  for (int i = tid; i < 32 * 6; i += 128) {
    const int r = i / 6, ch = 18 + (i % 6);
    *(v4bf*)&Ks[r * KS + ch * 4] = vz;
  }
  // Stage Q via async global->LDS B128 copies (9 x 16B per row)
  for (int i = tid; i < 64 * 9; i += 128) {
    const int r = i / 9, ch = i % 9;
    async_ld_b128(qbase + (size_t)r * (NH * DK) + ch * 8, &Qs[r * QS + ch * 8]);
  }
  wait_async0();
  __syncthreads();

  v16bf qa[3];
#pragma unroll
  for (int kk = 0; kk < 3; ++kk)
    qa[kk] = frag_a16(&Qs[(wave * 16) * QS + kk * 32], QS);

  v8f o[NVT] = {};
  float m_i[8], l_i[8];
#pragma unroll
  for (int r = 0; r < 8; ++r) { m_i[r] = -1e30f; l_i[r] = 0.0f; }

  for (int t0 = 0; t0 < T; t0 += 32) {
    __syncthreads();  // previous chunk's compute done before overwriting Ks/VsT
    // K chunk via async copies; token tail clamped (finite real data; scores overridden later)
    for (int i = tid; i < 32 * 9; i += 128) {
      const int r = i / 9, ch = i % 9;
      const int tc = min(t0 + r, T - 1);
      async_ld_b128(kbase + (size_t)tc * (NH * DK) + ch * 8, &Ks[r * KS + ch * 8]);
    }
    // V chunk -> transposed [feat][token]; token tail must be exact zeros (P mixes all tokens)
#pragma unroll
    for (int i = 0; i < (32 * NCHV) / 128; ++i) {
      const int idx = tid + i * 128;
      const int r = idx / NCHV, ch = idx % NCHV;
      const int tc = min(t0 + r, T - 1);
      const int chc = min(ch, DV / 4 - 1);
      v4bf v = *(const v4bf*)(vbase + (size_t)tc * (NH * DV) + chc * 4);
      if ((t0 + r >= T) || (ch >= DV / 4)) v = vz;
#pragma unroll
      for (int e = 0; e < 4; ++e)
        VsT[(ch * 4 + e) * 32 + r] = v[e];
    }
    if (tid < 32) {
      float ma = 0.0f;
      if (MASKED) {
        const int t = t0 + tid;
        if (t < T) ma = (mask[(size_t)b * T + t] == 0) ? -10000.0f : 1.0f;
      }
      Ma[tid] = ma;
    }
    wait_async0();
    __syncthreads();

    // S = Q @ K^T  (two 16-token tiles; K^T in B layout = contiguous per-lane runs)
    v8f s0 = {}; v8f s1 = {};
#pragma unroll
    for (int kk = 0; kk < 3; ++kk) {
      const v16bf k0f = frag_b16(&Ks[kk * 32], KS);
      const v16bf k1f = frag_b16(&Ks[16 * KS + kk * 32], KS);
      s0 = WMMA_BF16(qa[kk], k0f, s0);
      s1 = WMMA_BF16(qa[kk], k1f, s1);
    }

    const int cA = t0 + (lane & 15);
    const int cB = cA + 16;
    const float maA = Ma[lane & 15];
    const float maB = Ma[16 + (lane & 15)];
    float p0[8], p1[8], corr[8];
#pragma unroll
    for (int r = 0; r < 8; ++r) {
      float x0 = (cA < T) ? (s0[r] * scale + maA) : -1e30f;
      float x1 = (cB < T) ? (s1[r] * scale + maB) : -1e30f;
      float mx = fmaxf(x0, x1);
#pragma unroll
      for (int off = 1; off < 16; off <<= 1) mx = fmaxf(mx, __shfl_xor(mx, off, 32));
      const float mnew = fmaxf(m_i[r], mx);
      const float cfac = __expf(m_i[r] - mnew);
      const float e0 = __expf(x0 - mnew);
      const float e1 = __expf(x1 - mnew);
      float sm = e0 + e1;
#pragma unroll
      for (int off = 1; off < 16; off <<= 1) sm += __shfl_xor(sm, off, 32);
      l_i[r] = l_i[r] * cfac + sm;
      m_i[r] = mnew;
      p0[r] = e0; p1[r] = e1; corr[r] = cfac;
    }

#pragma unroll
    for (int j = 0; j < NVT; ++j)
#pragma unroll
      for (int r = 0; r < 8; ++r) o[j][r] = o[j][r] * corr[r];

    // Round-trip P through per-wave LDS tile to re-form A-fragment layout
    __bf16* Pw = &Ps[wave][0];
#pragma unroll
    for (int r = 0; r < 8; ++r) {
      const int prow = r + hi;
      Pw[prow * 40 + (lane & 15)]      = (__bf16)p0[r];
      Pw[prow * 40 + 16 + (lane & 15)] = (__bf16)p1[r];
    }
    const v16bf pa = frag_a16(Pw, 40);
#pragma unroll
    for (int j = 0; j < NVT; ++j) {
      const v16bf vb = frag_b16(&VsT[(j * 16) * 32], 32);
      o[j] = WMMA_BF16(pa, vb, o[j]);
    }
  }

  float* obase = Out + ((size_t)b * 64 + wave * 16) * (NH * DV) + h * DV;
#pragma unroll
  for (int r = 0; r < 8; ++r) {
    const float inv = 1.0f / l_i[r];
    const int row = r + hi;
#pragma unroll
    for (int j = 0; j < NVT; ++j) {
      const int d = j * 16 + (lane & 15);
      if (d < DV) obase[(size_t)row * (NH * DV) + d] = o[j][r] * inv;
    }
  }
}

// ---------------- Host orchestration ----------------

static inline void launch_gemm(const float* A, const float* W, const float* bias,
                               const float* resid, void* C, int M, int N, int K,
                               int out_bf16, hipStream_t s)
{
  dim3 grid(N / 64, (M + 127) / 128);
  if (out_bf16)    gemm_kernel<1, 0><<<grid, 256, 0, s>>>(A, W, bias, nullptr, C, M, N, K);
  else if (resid)  gemm_kernel<0, 1><<<grid, 256, 0, s>>>(A, W, bias, resid,  C, M, N, K);
  else             gemm_kernel<0, 0><<<grid, 256, 0, s>>>(A, W, bias, nullptr, C, M, N, K);
}

extern "C" void kernel_launch(void* const* d_in, const int* in_sizes, int n_in,
                              void* d_out, int out_size, void* d_ws, size_t ws_size,
                              hipStream_t stream)
{
  (void)in_sizes; (void)n_in; (void)out_size; (void)ws_size;

  const float* lt    = (const float*)d_in[0];   // [32,64,2560]
  const float* img   = (const float*)d_in[1];   // [32,729,1152]
  const float* txt   = (const float*)d_in[2];   // [32,512,2560]
  const int*   msk   = (const int*)  d_in[3];   // [32,512]
  const float* Wd_lt = (const float*)d_in[4];
  const float* bd_lt = (const float*)d_in[5];
  const float* Wd_im = (const float*)d_in[6];
  const float* bd_im = (const float*)d_in[7];
  const float* lnq_g = (const float*)d_in[8];
  const float* lnq_b = (const float*)d_in[9];
  const float* lnt_g = (const float*)d_in[10];
  const float* lnt_b = (const float*)d_in[11];
  const float* lnk_g = (const float*)d_in[12];
  const float* lnk_b = (const float*)d_in[13];
  const float* qW  = (const float*)d_in[14];
  const float* qb  = (const float*)d_in[15];
  const float* kW  = (const float*)d_in[16];
  const float* kb_ = (const float*)d_in[17];
  const float* vW  = (const float*)d_in[18];
  const float* vb  = (const float*)d_in[19];
  const float* q2W = (const float*)d_in[20];
  const float* q2b = (const float*)d_in[21];
  const float* k2W = (const float*)d_in[22];
  const float* k2b = (const float*)d_in[23];
  const float* v2W = (const float*)d_in[24];
  const float* v2b = (const float*)d_in[25];
  const float* upW = (const float*)d_in[26];
  const float* upb = (const float*)d_in[27];

  constexpr int MQ = 32 * 64;    // 2048
  constexpr int MT = 32 * 512;   // 16384
  constexpr int MV = 32 * 729;   // 23328

  char* ws = (char*)d_ws;
  constexpr size_t o_lt = 0;
  constexpr size_t o_tt = o_lt + (size_t)MQ * 576 * 4;
  constexpr size_t o_it = o_tt + (size_t)MT * 576 * 4;
  constexpr size_t o_qp = o_it + (size_t)MV * 576 * 4;
  constexpr size_t o_kp = o_qp + (size_t)MQ * 576 * 2;
  constexpr size_t o_vp = o_kp + (size_t)MT * 576 * 2;
  constexpr size_t o_x1 = o_vp + (size_t)MT * 576 * 2;
  constexpr size_t o_x2 = o_x1 + (size_t)MQ * 576 * 4;

  float*  lt_down = (float*)(ws + o_lt);
  float*  tt_down = (float*)(ws + o_tt);
  float*  it_down = (float*)(ws + o_it);
  __bf16* qp      = (__bf16*)(ws + o_qp);
  __bf16* kp      = (__bf16*)(ws + o_kp);
  __bf16* vp      = (__bf16*)(ws + o_vp);
  float*  x1      = (float*)(ws + o_x1);
  float*  x2      = (float*)(ws + o_x2);
  // Reused regions (previous occupants dead by the time these are written):
  __bf16* qp2 = (__bf16*)(ws + o_qp);  // over qp
  __bf16* kp2 = (__bf16*)(ws + o_kp);  // over kp+vp (26.9MB < 37.8MB)
  __bf16* vp2 = (__bf16*)(ws + o_lt);  // over lt/tt/it prefix (after kp2 GEMM reads it_down)

  const float scale = 0.11785113019775793f;  // 1/sqrt(72)

  // Down-projections
  launch_gemm(lt,  Wd_lt, bd_lt, nullptr, lt_down, MQ, 576, 2560, 0, stream);
  launch_gemm(txt, Wd_lt, bd_lt, nullptr, tt_down, MT, 576, 2560, 0, stream);
  launch_gemm(img, Wd_im, bd_im, nullptr, it_down, MV, 576, 1152, 0, stream);

  // LayerNorms (in place)
  ln_kernel<<<MQ, 256, 0, stream>>>(lt_down, lnq_g, lnq_b, 576);
  ln_kernel<<<MT, 256, 0, stream>>>(tt_down, lnt_g, lnt_b, 576);
  ln_kernel<<<MV, 256, 0, stream>>>(it_down, lnk_g, lnk_b, 576);

  // t2q projections (bf16 outputs for attention)
  launch_gemm(lt_down, qW, qb,  nullptr, qp, MQ, 576, 576, 1, stream);
  launch_gemm(tt_down, kW, kb_, nullptr, kp, MT, 576, 576, 1, stream);
  launch_gemm(tt_down, vW, vb,  nullptr, vp, MT, 576, 576, 1, stream);

  // Attention 1: dv=72, masked, T=512
  attn_kernel<72, true><<<dim3(32 * 8), 128, 0, stream>>>(qp, kp, vp, msk, x1, 512, scale);

  // i2q projections
  launch_gemm(x1,      q2W, q2b, nullptr, qp2, MQ, 576,  576,  1, stream);
  launch_gemm(it_down, k2W, k2b, nullptr, kp2, MV, 576,  576,  1, stream);
  launch_gemm(img,     v2W, v2b, nullptr, vp2, MV, 1152, 1152, 1, stream);

  // Attention 2: dv=144, unmasked, T=729
  attn_kernel<144, false><<<dim3(32 * 8), 128, 0, stream>>>(qp2, kp2, vp2, nullptr, x2, 729, scale);

  // Up-projection + residual -> d_out (f32)
  launch_gemm(x2, upW, upb, lt, (float*)d_out, MQ, 2560, 1152, 0, stream);
}
